// AttnBlock_78932908966272
// MI455X (gfx1250) — matrix operand order repaired and checked by reading, not compiled, not run on hardware
//
#include <hip/hip_runtime.h>
#include <hip/hip_bf16.h>
#include <math.h>

// ---------------------------------------------------------------------------
// AttnBlock: GN -> q,k,v (1x1 conv) -> softmax(q^T k / sqrt(C)) v -> proj -> +x
// All 6 GEMM phases run on v_wmma_f32_16x16x32_f16 (wave32 CDNA5 WMMA).
// Layouts are pre-arranged so WMMA A tiles are row-major contiguous and
// B tiles are column-contiguous => all operand loads are 16B global_load_b128.
// Each wave computes a 64x64 output block: 16 WMMAs per k-step vs 16 b128
// loads (1.0 loads/WMMA), letting each loadcnt wait cover 4 WMMAs.
// ---------------------------------------------------------------------------

typedef _Float16 v16h __attribute__((ext_vector_type(16)));
typedef _Float16 v8h  __attribute__((ext_vector_type(8)));
typedef float    v8f  __attribute__((ext_vector_type(8)));

#define NB  32          // batch
#define C_  512         // channels
#define HW_ 1024        // h*w
#define G_  32          // groups
#define TM  4           // 16-row tiles per wave (M block = 64)
#define TN  4           // 16-col tiles per wave (N block = 64)

// ---- WMMA operand loaders (per ISA 7.12.2 VGPR layouts, wave32) -----------

// A tile: 16x32 (MxK), row-major with leading dim ld (elements).
// lane L: m = L&15, khalf = L>>4. VGPR0-3 hold K = khalf*8 + 0..7,
// VGPR4-7 hold K = 16 + khalf*8 + 0..7  => two 16B loads per lane.
__device__ inline v16h load_tileA(const _Float16* __restrict__ base, int ld) {
  const int lane = threadIdx.x & 31;
  const int m    = lane & 15;
  const int kh   = (lane >> 4) & 1;
  const _Float16* p = base + (long long)m * ld + kh * 8;
  v8h lo = *(const v8h*)(p);
  v8h hi = *(const v8h*)(p + 16);
  v16h r;
#pragma unroll
  for (int i = 0; i < 8; ++i) { r[i] = lo[i]; r[i + 8] = hi[i]; }
  return r;
}

// B tile: 32x16 (KxN), stored column-contiguous: Bt[n*ld + k] == B[k][n].
// lane L: n = L&15, khalf = L>>4; lane holds B[khalf*8 + 0..7][n] and
// B[16 + khalf*8 + 0..7][n] => two 16B loads per lane.
__device__ inline v16h load_tileBt(const _Float16* __restrict__ base, int ld) {
  const int lane = threadIdx.x & 31;
  const int n    = lane & 15;
  const int kh   = (lane >> 4) & 1;
  const _Float16* p = base + (long long)n * ld + kh * 8;
  v8h lo = *(const v8h*)(p);
  v8h hi = *(const v8h*)(p + 16);
  v16h r;
#pragma unroll
  for (int i = 0; i < 8; ++i) { r[i] = lo[i]; r[i + 8] = hi[i]; }
  return r;
}

// ---- Generic batched WMMA GEMM: D = scale*(A x B) + bias ------------------
// A: row-major MxK (lda), Bt: column-contiguous (ldb = K-dim leading),
// D: row-major MxN (ldd), f32 or f16 output. biasMode: 0 none, 1 per-col(n),
// 2 per-row(m). Each wave computes a 64x64 block (16 WMMAs per k-step).
__global__ __launch_bounds__(256) void wmma_gemm_kernel(
    const _Float16* __restrict__ A,  long long strideA, int lda,
    const _Float16* __restrict__ Bt, long long strideB, int ldb,
    float* __restrict__ Df32, _Float16* __restrict__ Df16,
    long long strideD, int ldd,
    const float* __restrict__ bias, int biasMode, float scale,
    int M, int N, int K, int nBatch)
{
  const int wave    = blockIdx.x * (blockDim.x >> 5) + (threadIdx.x >> 5);
  const int mBlocks = M / (16 * TM);
  const int nBlocks = N / (16 * TN);
  const int wpb     = mBlocks * nBlocks;       // waves per batch
  const int batch   = wave / wpb;
  if (batch >= nBatch) return;
  const int rem = wave - batch * wpb;
  const int m0  = (rem / nBlocks) * (16 * TM);
  const int n0  = (rem % nBlocks) * (16 * TN);

  const _Float16* Ab = A  + (long long)batch * strideA + (long long)m0 * lda;
  const _Float16* Bb = Bt + (long long)batch * strideB + (long long)n0 * ldb;

  v8f acc[TM][TN];
#pragma unroll
  for (int mt = 0; mt < TM; ++mt)
#pragma unroll
    for (int nt = 0; nt < TN; ++nt) acc[mt][nt] = (v8f){};

  for (int k = 0; k < K; k += 32) {
    v16h bt[TN];
#pragma unroll
    for (int nt = 0; nt < TN; ++nt)
      bt[nt] = load_tileBt(Bb + (long long)(nt * 16) * ldb + k, ldb);
#pragma unroll
    for (int mt = 0; mt < TM; ++mt) {
      v16h at = load_tileA(Ab + (long long)(mt * 16) * lda + k, lda);
#pragma unroll
      for (int nt = 0; nt < TN; ++nt)
        acc[mt][nt] = __builtin_amdgcn_wmma_f32_16x16x32_f16(
            false, at, false, bt[nt], (short)0, acc[mt][nt], false, false);
    }
  }

  // Epilogue: C/D layout -> lane L, VGPR r holds D[(L>>4)*8 + r][L&15]
  const int lane = threadIdx.x & 31;
  const int nl   = lane & 15;
  const int mh   = (lane >> 4) & 1;
  if (Df16) {
#pragma unroll
    for (int mt = 0; mt < TM; ++mt)
#pragma unroll
      for (int nt = 0; nt < TN; ++nt)
#pragma unroll
        for (int r = 0; r < 8; ++r) {
          const int m = m0 + mt * 16 + mh * 8 + r;
          const int n = n0 + nt * 16 + nl;
          float v = acc[mt][nt][r] * scale;
          if (biasMode == 1)      v += bias[n];
          else if (biasMode == 2) v += bias[m];
          Df16[(long long)batch * strideD + (long long)m * ldd + n] = (_Float16)v;
        }
  } else {
#pragma unroll
    for (int mt = 0; mt < TM; ++mt)
#pragma unroll
      for (int nt = 0; nt < TN; ++nt)
#pragma unroll
        for (int r = 0; r < 8; ++r) {
          const int m = m0 + mt * 16 + mh * 8 + r;
          const int n = n0 + nt * 16 + nl;
          float v = acc[mt][nt][r] * scale;
          if (biasMode == 1)      v += bias[n];
          else if (biasMode == 2) v += bias[m];
          Df32[(long long)batch * strideD + (long long)m * ldd + n] = v;
        }
  }
}

// ---- GroupNorm: per (b,g) mean/var over 16ch x 1024, write hnT (HW x C) ---
__global__ __launch_bounds__(256) void groupnorm_kernel(
    const float* __restrict__ x, const float* __restrict__ gamma,
    const float* __restrict__ beta, _Float16* __restrict__ hnT)
{
  const int bg = blockIdx.x;             // b*G + g
  const int b  = bg >> 5;
  const int g  = bg & 31;
  const int c0 = g * (C_ / G_);          // 16 channels per group
  const float* xb = x + ((long long)b * C_ + c0) * HW_;
  const int NE = (C_ / G_) * HW_;        // 16384

  float s = 0.f, ss = 0.f;
  for (int e = threadIdx.x; e < NE; e += 256) {
    float v = xb[e]; s += v; ss += v * v;
  }
  __shared__ float red[512];
  red[threadIdx.x] = s;
  red[256 + threadIdx.x] = ss;
  __syncthreads();
  for (int off = 128; off > 0; off >>= 1) {
    if (threadIdx.x < off) {
      red[threadIdx.x]       += red[threadIdx.x + off];
      red[256 + threadIdx.x] += red[256 + threadIdx.x + off];
    }
    __syncthreads();
  }
  if (threadIdx.x == 0) {
    float mean = red[0] / NE;
    float var  = red[256] / NE - mean * mean;
    red[0] = mean;
    red[1] = rsqrtf(var + 1e-5f);
  }
  __syncthreads();
  const float mean = red[0], rstd = red[1];
  for (int e = threadIdx.x; e < NE; e += 256) {
    const int ch = e >> 10;
    const int p  = e & (HW_ - 1);
    const int c  = c0 + ch;
    float v = (xb[e] - mean) * rstd * gamma[c] + beta[c];
    hnT[((long long)b * HW_ + p) * C_ + c] = (_Float16)v;
  }
}

// ---- Softmax: one wave32 per 1024-wide row, shfl_xor reductions -----------
__global__ __launch_bounds__(256) void softmax_kernel(
    const float* __restrict__ S, _Float16* __restrict__ attn)
{
  const int row  = blockIdx.x * 8 + (threadIdx.x >> 5);  // b*HW + p
  const int lane = threadIdx.x & 31;
  const float* srow = S + (long long)row * HW_;

  float vals[32];
  float mx = -INFINITY;
#pragma unroll
  for (int i = 0; i < 32; ++i) {
    vals[i] = srow[lane + i * 32];
    mx = fmaxf(mx, vals[i]);
  }
#pragma unroll
  for (int off = 16; off > 0; off >>= 1) mx = fmaxf(mx, __shfl_xor(mx, off, 32));
  float sum = 0.f;
#pragma unroll
  for (int i = 0; i < 32; ++i) { vals[i] = __expf(vals[i] - mx); sum += vals[i]; }
#pragma unroll
  for (int off = 16; off > 0; off >>= 1) sum += __shfl_xor(sum, off, 32);
  const float inv = 1.0f / sum;
  _Float16* arow = attn + (long long)row * HW_;
#pragma unroll
  for (int i = 0; i < 32; ++i) arow[lane + i * 32] = (_Float16)(vals[i] * inv);
}

// ---- Residual: out[b][c][p] = x + yT[b][p][c] + bp[c] ---------------------
__global__ __launch_bounds__(256) void residual_kernel(
    const float* __restrict__ x, const float* __restrict__ yT,
    const float* __restrict__ bp, float* __restrict__ out)
{
  const long long i = (long long)blockIdx.x * 256 + threadIdx.x;
  const int p = (int)(i & (HW_ - 1));
  const long long t = i >> 10;          // b*C + c
  const int c = (int)(t & (C_ - 1));
  const long long b = t >> 9;
  out[i] = x[i] + yT[(b * HW_ + p) * C_ + c] + bp[c];
}

// ---- f32 -> f16 weight conversion -----------------------------------------
__global__ __launch_bounds__(256) void cvt_f16_kernel(
    const float* __restrict__ src, _Float16* __restrict__ dst, int n)
{
  for (int i = blockIdx.x * 256 + threadIdx.x; i < n; i += gridDim.x * 256)
    dst[i] = (_Float16)src[i];
}

// ---------------------------------------------------------------------------
extern "C" void kernel_launch(void* const* d_in, const int* in_sizes, int n_in,
                              void* d_out, int out_size, void* d_ws, size_t ws_size,
                              hipStream_t stream) {
  const float* x  = (const float*)d_in[0];
  const float* gs = (const float*)d_in[1];
  const float* gb = (const float*)d_in[2];
  const float* wq = (const float*)d_in[3];
  const float* bq = (const float*)d_in[4];
  const float* wk = (const float*)d_in[5];
  const float* bk = (const float*)d_in[6];
  const float* wv = (const float*)d_in[7];
  const float* bv = (const float*)d_in[8];
  const float* wp = (const float*)d_in[9];
  const float* bp = (const float*)d_in[10];
  float* out = (float*)d_out;

  // Workspace layout (all offsets 256B aligned). Total ~354 MB.
  char* ws = (char*)d_ws;
  size_t off = 0;
  auto take = [&](size_t bytes) {
    char* p = ws + off;
    off += (bytes + 255) & ~(size_t)255;
    return p;
  };
  const size_t SZ_HWC = (size_t)NB * HW_ * C_;       // 16.7M elements
  _Float16* hnT   = (_Float16*)take(SZ_HWC * 2);     // GN out  (B,HW,C)
  _Float16* qT    = (_Float16*)take(SZ_HWC * 2);     // (B,HW,C)
  _Float16* kT    = (_Float16*)take(SZ_HWC * 2);     // (B,HW,C)
  _Float16* vC    = (_Float16*)take(SZ_HWC * 2);     // (B,C,HW)
  _Float16* oT    = (_Float16*)take(SZ_HWC * 2);     // (B,HW,C)
  _Float16* attn  = (_Float16*)take((size_t)NB * HW_ * HW_ * 2); // (B,HW,HW)
  _Float16* wq16  = (_Float16*)take((size_t)C_ * C_ * 2);
  _Float16* wk16  = (_Float16*)take((size_t)C_ * C_ * 2);
  _Float16* wv16  = (_Float16*)take((size_t)C_ * C_ * 2);
  _Float16* wp16  = (_Float16*)take((size_t)C_ * C_ * 2);
  float*    Sbuf  = (float*)take((size_t)NB * HW_ * HW_ * 4);    // scores f32
  float*    yT    = Sbuf;   // reused after softmax (64MB <= 128MB)

  const int nW = C_ * C_;   // 262144 weight elements each
  cvt_f16_kernel<<<256, 256, 0, stream>>>(wq, wq16, nW);
  cvt_f16_kernel<<<256, 256, 0, stream>>>(wk, wk16, nW);
  cvt_f16_kernel<<<256, 256, 0, stream>>>(wv, wv16, nW);
  cvt_f16_kernel<<<256, 256, 0, stream>>>(wp, wp16, nW);

  groupnorm_kernel<<<NB * G_, 256, 0, stream>>>(x, gs, gb, hnT);

  auto gemm = [&](const _Float16* A, long long sA, int lda,
                  const _Float16* Bt, long long sB, int ldb,
                  float* D32, _Float16* D16, long long sD, int ldd,
                  const float* bias, int bm, float scale,
                  int M, int N, int K) {
    const int waves = NB * (M / (16 * TM)) * (N / (16 * TN));
    wmma_gemm_kernel<<<waves / 8, 256, 0, stream>>>(
        A, sA, lda, Bt, sB, ldb, D32, D16, sD, ldd, bias, bm, scale, M, N, K, NB);
  };

  const long long sHWC = (long long)HW_ * C_;
  const long long sHH  = (long long)HW_ * HW_;
  const float SCL = 0.044194173824159216f;  // 512^-0.5

  // q,k: (HW x C) = hnT (HW x C) x Wq^T  ; bias per output channel (col)
  gemm(hnT, sHWC, C_, wq16, 0, C_, nullptr, qT, sHWC, C_, bq, 1, 1.f, HW_, C_, C_);
  gemm(hnT, sHWC, C_, wk16, 0, C_, nullptr, kT, sHWC, C_, bk, 1, 1.f, HW_, C_, C_);
  // v: (C x HW) = Wv (C x C) x hn ; B^T = hnT ; bias per row (out channel)
  gemm(wv16, 0, C_, hnT, sHWC, C_, nullptr, vC, sHWC, HW_, bv, 2, 1.f, C_, HW_, C_);
  // scores: (HW x HW) = qT x k ; B^T = kT ; scaled, f32 out
  gemm(qT, sHWC, C_, kT, sHWC, C_, Sbuf, nullptr, sHH, HW_, nullptr, 0, SCL,
       HW_, HW_, C_);

  softmax_kernel<<<NB * HW_ / 8, 256, 0, stream>>>(Sbuf, attn);

  // oT: (HW x C) = attn (HW x HW) x v^T ; B^T = vC
  gemm(attn, sHH, HW_, vC, sHWC, HW_, nullptr, oT, sHWC, C_, nullptr, 0, 1.f,
       HW_, C_, HW_);
  // yT: (HW x C) = oT x Wp^T ; f32 out into reused scores buffer
  gemm(oT, sHWC, C_, wp16, 0, C_, yT, nullptr, sHWC, C_, nullptr, 0, 1.f,
       HW_, C_, C_);

  const long long total = (long long)NB * C_ * HW_;
  residual_kernel<<<(int)(total / 256), 256, 0, stream>>>(x, yT, bp, out);
}
